// RingConv2dCUDA_18476949307605
// MI455X (gfx1250) — matrix operand order repaired and checked by reading, not compiled, not run on hardware
//
#include <hip/hip_runtime.h>
#include <hip/hip_bf16.h>

typedef __attribute__((ext_vector_type(16))) _Float16 v16h;
typedef __attribute__((ext_vector_type(8)))  float    v8f;

#define CIN   128
#define COUT  256
#define C2    256               // 2*CIN  (cos-channels ; sin-channels)
#define TAPS  9
#define KTOT  (C2 * TAPS)       // 2304, k = tap*256 + c2  (tap-major)
#define BATCH 16
#define H     56
#define W     56
#define HW    (H * W)
#define NPIX  (BATCH * HW)      // 50176 = 784 * 64
#define M_BLK 128
#define N_BLK 64
#define K_STEP 32
#define NIT   (KTOT / K_STEP)   // 72

// ---------------------------------------------------------------------------
// CDNA5 async copy helpers: global -> LDS via the async engine (ASYNCcnt).
// INST_OFFSET is added to BOTH the LDS and global addresses (ISA §10.7/15.18),
// so one address pair serves two 16B beats.
// ---------------------------------------------------------------------------
__device__ __forceinline__ void async_b128(unsigned lds_off, const void* gaddr) {
  asm volatile("global_load_async_to_lds_b128 %0, %1, off"
               :: "v"(lds_off), "v"(gaddr) : "memory");
}
__device__ __forceinline__ void async_b128_o16(unsigned lds_off, const void* gaddr) {
  asm volatile("global_load_async_to_lds_b128 %0, %1, off offset:16"
               :: "v"(lds_off), "v"(gaddr) : "memory");
}
__device__ __forceinline__ unsigned lds_off32(const void* p) {
  // generic LDS pointer: addr[31:0] is the LDS byte address (ISA §10.2)
  return (unsigned)(uintptr_t)p;
}

// ---------------------------------------------------------------------------
// Prep 1: trig-product weights -> f16, row-major [m=512][k=2304].
// ---------------------------------------------------------------------------
__global__ void prep_weights(const float* __restrict__ probe,
                             const float* __restrict__ outang,
                             _Float16* __restrict__ wgt) {
  int idx = blockIdx.x * blockDim.x + threadIdx.x;
  if (idx >= 512 * KTOT) return;
  int m   = idx / KTOT;
  int k   = idx - m * KTOT;
  int tap = k >> 8;           // k = tap*256 + c2
  int c2  = k & 255;
  int cout = m & 255;
  int cin  = c2 & 127;
  int pi = (cin * COUT + cout) * TAPS + tap;   // (1,Cin,Cout,1,1,3,3) flat
  float p = probe[pi], o = outang[pi];
  float wp = (c2 >= CIN) ? __sinf(p) : __cosf(p);
  float wo = (m >= 256)  ? __sinf(o) : __cosf(o);
  wgt[m * KTOT + k] = (_Float16)(wp * wo);
}

// ---------------------------------------------------------------------------
// Prep 2: input -> NHWC f16  cs[b][h][w][c2], c2<128 = cos(x), else sin(x).
// ---------------------------------------------------------------------------
__global__ void prep_input(const float* __restrict__ x,
                           _Float16* __restrict__ cs) {
  int idx = blockIdx.x * blockDim.x + threadIdx.x;
  if (idx >= NPIX * C2) return;
  int c2 = idx & 255;
  int p  = idx >> 8;
  int cin = c2 & 127;
  int b   = p / HW;
  int hw  = p - b * HW;
  float v = x[(b * CIN + cin) * HW + hw];
  cs[p * C2 + c2] = (_Float16)((c2 < CIN) ? __cosf(v) : __sinf(v));
}

// ---------------------------------------------------------------------------
// Implicit-GEMM ring conv with double-buffered async-to-LDS staging.
// Block = 256 threads (8 waves), tile 128M x 64N; wave tile 32x32 with
// re+im accumulator sets so atan2 fuses in-register.
// ---------------------------------------------------------------------------
__global__ __launch_bounds__(256)
void ring_gemm(const _Float16* __restrict__ wgt,
               const _Float16* __restrict__ cs,
               const void* __restrict__ zp,      // 64B zero page for padding
               float* __restrict__ y) {
  __shared__ __align__(16) _Float16 ldsAre[2][M_BLK * K_STEP];  // 2 x 8 KB
  __shared__ __align__(16) _Float16 ldsAim[2][M_BLK * K_STEP];  // 2 x 8 KB
  __shared__ __align__(16) _Float16 ldsB  [2][N_BLK * K_STEP];  // 2 x 4 KB

  const int tid  = threadIdx.x;
  const int lane = tid & 31;
  const int wave = tid >> 5;
  const int mw   = (wave >> 1) * 32;
  const int nw   = (wave & 1)  * 32;
  const int m0   = blockIdx.y * M_BLK;
  const int n0   = blockIdx.x * N_BLK;

  v8f acc_re[2][2] = {};
  v8f acc_im[2][2] = {};

  // A staging: thread -> (row, 32B half of 64B row); 4 async b128 / thread
  const int arow  = tid >> 1;
  const int abyte = (tid & 1) * 32;
  // B staging: thread -> (pixel, 16B segment of 64B run); 1 async b128 / thread
  const int bpix  = tid >> 2;
  const int bseg  = (tid & 3) * 16;
  const int pix   = n0 + bpix;
  const int pb    = pix / HW;
  const int prem  = pix - pb * HW;
  const int pho   = prem / W;
  const int pwo   = prem - pho * W;

  const int lrow = lane & 15;
  const int lgrp = lane >> 4;

  const _Float16* gre_base = wgt + (size_t)(m0 + arow) * KTOT + (abyte >> 1);
  const _Float16* gim_base = gre_base + (size_t)256 * KTOT;

  auto issue = [&](int k0, int p) {
    // ---- A tiles: always in range ----
    const _Float16* gre = gre_base + k0;
    const _Float16* gim = gim_base + k0;
    unsigned lre = lds_off32(&ldsAre[p][0]) + arow * 64 + abyte;
    unsigned lim = lds_off32(&ldsAim[p][0]) + arow * 64 + abyte;
    async_b128(lre, gre);  async_b128_o16(lre, gre);
    async_b128(lim, gim);  async_b128_o16(lim, gim);
    // ---- B tile: im2col gather, OOB taps redirected to zero page ----
    const int tap  = k0 >> 8;
    const int c2_0 = k0 & 255;
    const int kh = (tap * 11) >> 5;          // tap/3 for tap<9
    const int kw = tap - 3 * kh;
    const int hi = pho + kh - 1, wi = pwo + kw - 1;
    const void* gB = (hi >= 0 && hi < H && wi >= 0 && wi < W)
        ? (const void*)(cs + ((size_t)((pb * H + hi) * W + wi)) * C2 + c2_0 + (bseg >> 1))
        : zp;
    unsigned lB = lds_off32(&ldsB[p][0]) + bpix * 64 + bseg;
    async_b128(lB, gB);
    __builtin_prefetch(gre + 2 * K_STEP, 0, 1);   // K+2 chunk hint
  };

  issue(0, 0);
  for (int it = 0; it < NIT; ++it) {
    const int p = it & 1;
    if (it + 1 < NIT) {
      issue((it + 1) * K_STEP, p ^ 1);
      // in-order ASYNC retirement: <=5 outstanding => chunk `it` landed
      asm volatile("s_wait_asynccnt 0x5" ::: "memory");
    } else {
      asm volatile("s_wait_asynccnt 0x0" ::: "memory");
    }
    __syncthreads();   // chunk `it` visible to all waves

    // ---- fragments per ISA 16-bit layouts ----
    v16h afr[2], afi[2], bf[2];
#pragma unroll
    for (int i = 0; i < 2; ++i) {
      const int r = mw + i * 16 + lrow;
      const char* br = (const char*)&ldsAre[p][0] + r * 64 + lgrp * 16;
      ((uint4*)&afr[i])[0] = *(const uint4*)(br);
      ((uint4*)&afr[i])[1] = *(const uint4*)(br + 32);
      const char* bi = (const char*)&ldsAim[p][0] + r * 64 + lgrp * 16;
      ((uint4*)&afi[i])[0] = *(const uint4*)(bi);
      ((uint4*)&afi[i])[1] = *(const uint4*)(bi + 32);
    }
#pragma unroll
    for (int j = 0; j < 2; ++j) {
      const int c = nw + j * 16 + lrow;
      const char* bb = (const char*)&ldsB[p][0] + c * 64 + lgrp * 16;
      ((uint4*)&bf[j])[0] = *(const uint4*)(bb);
      ((uint4*)&bf[j])[1] = *(const uint4*)(bb + 32);
    }

#pragma unroll
    for (int i = 0; i < 2; ++i)
#pragma unroll
      for (int j = 0; j < 2; ++j) {
        acc_re[i][j] = __builtin_amdgcn_wmma_f32_16x16x32_f16(
            false, afr[i], false, bf[j], (short)0, acc_re[i][j], false, false);
        acc_im[i][j] = __builtin_amdgcn_wmma_f32_16x16x32_f16(
            false, afi[i], false, bf[j], (short)0, acc_im[i][j], false, false);
      }
    __syncthreads();   // all waves done reading buffer p before it is refilled
  }

  // ---- epilogue: fused atan2, scatter to NCHW ----
#pragma unroll
  for (int i = 0; i < 2; ++i) {
    const int mbase = m0 + mw + i * 16 + lgrp * 8;
#pragma unroll
    for (int j = 0; j < 2; ++j) {
      const int px = n0 + nw + j * 16 + lrow;
      const int ob = px / HW;
      const int orem = px - ob * HW;
#pragma unroll
      for (int g = 0; g < 8; ++g) {
        const int cout = mbase + g;
        const float re = acc_re[i][j][g];
        const float im = acc_im[i][j][g];
        y[(size_t)(ob * COUT + cout) * HW + orem] = atan2f(im, re);
      }
    }
  }
}

// ---------------------------------------------------------------------------
extern "C" void kernel_launch(void* const* d_in, const int* in_sizes, int n_in,
                              void* d_out, int out_size, void* d_ws, size_t ws_size,
                              hipStream_t stream) {
  (void)in_sizes; (void)n_in; (void)out_size; (void)ws_size;
  const float* x      = (const float*)d_in[0];
  const float* probe  = (const float*)d_in[1];
  const float* outang = (const float*)d_in[2];
  float* y = (float*)d_out;

  char* ws = (char*)d_ws;
  void*     zp  = ws;                                    // 64B zero page
  _Float16* wgt = (_Float16*)(ws + 256);                 // 512*2304*2B = 2.36 MB
  _Float16* cs  = wgt + (size_t)512 * KTOT;              // 50176*256*2B = 25.7 MB

  hipMemsetAsync(zp, 0, 64, stream);
  {
    int n = 512 * KTOT;
    prep_weights<<<(n + 255) / 256, 256, 0, stream>>>(probe, outang, wgt);
  }
  {
    int n = NPIX * C2;
    prep_input<<<(n + 255) / 256, 256, 0, stream>>>(x, cs);
  }
  dim3 grid(NPIX / N_BLK, COUT / M_BLK);                 // 784 x 2
  ring_gemm<<<grid, 256, 0, stream>>>(wgt, cs, zp, y);
}
